// MambaBlock3D_24988119728749
// MI455X (gfx1250) — matrix operand order repaired and checked by reading, not compiled, not run on hardware
//
#include <hip/hip_runtime.h>
#include <math.h>

// ---------------------------------------------------------------------------
// MambaBlock3D for gfx1250 (MI455X): WMMA f16 GEMMs + VALU selective scan.
// Shapes: B=2, Cin=32, H=W=32, Dep=16, Cout=64, d_inner=128, d_state=16,
// dt_rank=4, d_conv=3.  M = B*H*W*Dep = 32768 "token" rows throughout.
// ---------------------------------------------------------------------------

typedef __attribute__((ext_vector_type(16))) _Float16 v16h;
typedef __attribute__((ext_vector_type(8)))  _Float16 v8h;
typedef __attribute__((ext_vector_type(8)))  float    v8f;

__device__ __forceinline__ v16h ld8x2(const _Float16* p0, const _Float16* p1) {
  v8h a = *reinterpret_cast<const v8h*>(p0);
  v8h b = *reinterpret_cast<const v8h*>(p1);
  v16h r;
#pragma unroll
  for (int i = 0; i < 8; ++i) { r[i] = a[i]; r[i + 8] = b[i]; }
  return r;
}

__device__ __forceinline__ v8f wmma16(v16h a, v16h b, v8f c) {
  // D = A(16x32 f16) * B(32x16 f16) + C(16x16 f32)
  return __builtin_amdgcn_wmma_f32_16x16x32_f16(false, a, false, b, (short)0, c,
                                                false, false);
}

__device__ __forceinline__ v8f vzero8() {
  v8f r;
#pragma unroll
  for (int i = 0; i < 8; ++i) r[i] = 0.f;
  return r;
}

// ---------------------------------------------------------------------------
// K1: pack/convert weights + activations to f16, zero BN accumulators.
// ---------------------------------------------------------------------------
__global__ void k_pack(const float* __restrict__ x, const float* __restrict__ conv_w,
                       const float* __restrict__ in_proj_w, const float* __restrict__ x_proj_w,
                       const float* __restrict__ out_proj_w,
                       _Float16* __restrict__ x_h, _Float16* __restrict__ wc_h,
                       _Float16* __restrict__ w1_h, _Float16* __restrict__ w2_h,
                       _Float16* __restrict__ w3_h,
                       float* __restrict__ ch_sum, float* __restrict__ ch_sumsq) {
  const int i = blockIdx.x * 256 + threadIdx.x;          // covers 1048576
  if (i < 1048576) x_h[i] = (_Float16)x[i];              // (B,Cin,H,W,Dep)
  if (i < 55296) {                                       // wc_h: [tap][cout][cin]
    const int cin = i & 31;
    const int cout = (i >> 5) & 63;
    const int tap = i >> 11;
    wc_h[i] = (_Float16)conv_w[(cout * 32 + cin) * 27 + tap];
  }
  if (i < 16384) w1_h[i] = (_Float16)in_proj_w[i];       // (256,64) row-major N x K
  if (i < 6144) {                                        // x_proj padded to (48,128)
    const int row = i >> 7, col = i & 127;
    w2_h[i] = (row < 36) ? (_Float16)x_proj_w[row * 128 + col] : (_Float16)0.f;
  }
  if (i < 8192) w3_h[i] = (_Float16)out_proj_w[i];       // (64,128)
  if (i < 64) { ch_sum[i] = 0.f; ch_sumsq[i] = 0.f; }
}

// ---------------------------------------------------------------------------
// K2: Conv3d as implicit GEMM with WMMA. M=32768 positions, N=64 ch, K=27x32.
// Epilogue: write y in u-layout (bf, l, c) and accumulate BN sum/sumsq.
// Tiles: 2048 Mtiles x 4 Ntiles = 8192; 8 waves/block -> 1024 blocks exact.
// ---------------------------------------------------------------------------
__global__ void k_conv_wmma(const _Float16* __restrict__ x_h, const _Float16* __restrict__ wc_h,
                            float* __restrict__ y, float* __restrict__ ch_sum,
                            float* __restrict__ ch_sumsq) {
  const int lane = threadIdx.x & 31;
  const int tile = blockIdx.x * 8 + (threadIdx.x >> 5);
  const int mt = tile >> 2;            // 0..2047 == (b,h,w)
  const int nt = tile & 3;             // 0..3
  const int b = mt >> 10;
  const int hw = mt & 1023;
  const int h = hw >> 5;
  const int w = hw & 31;
  const int dep = lane & 15;           // row-in-tile == dep (Dep==16, mtile aligned)
  const int cin0 = (lane & 16) ? 8 : 0;
  const int col = nt * 16 + (lane & 15);

  v8f acc = vzero8();
  for (int tap = 0; tap < 27; ++tap) {
    const int dh = tap / 9, rem = tap % 9, dw = rem / 3, dd = rem % 3;
    const int hh = h + dh - 1, ww = w + dw - 1, dd2 = dep + dd - 1;
    v16h a;
#pragma unroll
    for (int i = 0; i < 16; ++i) a[i] = (_Float16)0.f;
    if (hh >= 0 && hh < 32 && ww >= 0 && ww < 32 && dd2 >= 0 && dd2 < 16) {
      const size_t base0 = ((((size_t)b * 32) * 32 + hh) * 32 + ww) * 16 + dd2;
#pragma unroll
      for (int i = 0; i < 8; ++i) a[i] = x_h[base0 + (size_t)(cin0 + i) * 16384];
#pragma unroll
      for (int i = 0; i < 8; ++i) a[i + 8] = x_h[base0 + (size_t)(cin0 + 16 + i) * 16384];
    }
    const _Float16* bp = wc_h + ((size_t)(tap * 64 + col)) * 32 + ((lane & 16) ? 8 : 0);
    if (tap + 1 < 27) __builtin_prefetch(bp + 2048, 0, 0);   // global_prefetch next tap
    v16h bm = ld8x2(bp, bp + 16);
    acc = wmma16(a, bm, acc);
  }

  const int rowoff = (lane & 16) ? 8 : 0;
  float s = 0.f, sq = 0.f;
#pragma unroll
  for (int r = 0; r < 8; ++r) {
    const int dep2 = rowoff + r;
    const int bf = b * 16 + dep2;
    const float v = acc[r];
    y[((size_t)bf * 1024 + hw) * 64 + col] = v;
    s += v; sq += v * v;
  }
  atomicAdd(&ch_sum[col], s);
  atomicAdd(&ch_sumsq[col], sq);
}

// ---------------------------------------------------------------------------
// K3: finalize BN per-channel scale/shift (batch statistics).
// ---------------------------------------------------------------------------
__global__ void k_bnfin(const float* __restrict__ ch_sum, const float* __restrict__ ch_sumsq,
                        const float* __restrict__ gamma, const float* __restrict__ beta,
                        float* __restrict__ scale, float* __restrict__ shift) {
  const int c = threadIdx.x;
  if (c < 64) {
    const float n = 32768.f;
    const float mu = ch_sum[c] / n;
    const float var = ch_sumsq[c] / n - mu * mu;
    const float sc = gamma[c] * rsqrtf(var + 1e-5f);
    scale[c] = sc;
    shift[c] = beta[c] - mu * sc;
  }
}

// ---------------------------------------------------------------------------
// K4: BN-apply + ReLU, produce f16 u (32768 x 64) for in_proj WMMA.
// ---------------------------------------------------------------------------
__global__ void k_bnrelu(const float* __restrict__ y, const float* __restrict__ scale,
                         const float* __restrict__ shift, _Float16* __restrict__ u_h) {
  const int i = blockIdx.x * 256 + threadIdx.x;   // < 2097152
  const int c = i & 63;
  const float v = y[i] * scale[c] + shift[c];
  u_h[i] = (_Float16)fmaxf(v, 0.f);
}

// ---------------------------------------------------------------------------
// K5: in_proj GEMM  xz[m,n] = sum_k u[m,k] * W1[n,k];  M=32768, N=256, K=64.
// Tiles 2048x16 = 32768 -> 4096 blocks of 8 waves.
// ---------------------------------------------------------------------------
__global__ void k_gemm_inproj(const _Float16* __restrict__ u_h, const _Float16* __restrict__ w1_h,
                              float* __restrict__ xi, float* __restrict__ z) {
  const int lane = threadIdx.x & 31;
  const int tile = blockIdx.x * 8 + (threadIdx.x >> 5);
  const int mt = tile >> 4;
  const int nt = tile & 15;
  const int mrow = mt * 16 + (lane & 15);
  const int col = nt * 16 + (lane & 15);
  v8f acc = vzero8();
#pragma unroll
  for (int kc = 0; kc < 2; ++kc) {
    const int kb = kc * 32;
    const _Float16* ap = u_h + (size_t)mrow * 64 + kb + ((lane & 16) ? 8 : 0);
    v16h a = ld8x2(ap, ap + 16);
    const _Float16* bp = w1_h + (size_t)col * 64 + kb + ((lane & 16) ? 8 : 0);
    v16h bm = ld8x2(bp, bp + 16);
    acc = wmma16(a, bm, acc);
  }
  const int rowbase = mt * 16 + ((lane & 16) ? 8 : 0);
#pragma unroll
  for (int r = 0; r < 8; ++r) {
    const int m = rowbase + r;
    const float v = acc[r];
    if (col < 128) xi[(size_t)m * 128 + col] = v;
    else           z[(size_t)m * 128 + (col - 128)] = v;
  }
}

// ---------------------------------------------------------------------------
// K6: causal depthwise conv1d (k=3) along l within each bf, + bias + SiLU.
// ---------------------------------------------------------------------------
__global__ void k_conv1d(const float* __restrict__ xi, const float* __restrict__ w,
                         const float* __restrict__ bias, float* __restrict__ xic,
                         _Float16* __restrict__ xih) {
  const int idx = blockIdx.x * 256 + threadIdx.x;   // < 4194304
  const int di = idx & 127;
  const int m = idx >> 7;
  const int l = m & 1023;
  const float x2 = xi[idx];
  const float x1 = (l >= 1) ? xi[idx - 128] : 0.f;
  const float x0 = (l >= 2) ? xi[idx - 256] : 0.f;
  float v = w[di * 3 + 0] * x0 + w[di * 3 + 1] * x1 + w[di * 3 + 2] * x2 + bias[di];
  v = v / (1.f + __expf(-v));                       // SiLU
  xic[idx] = v;
  xih[idx] = (_Float16)v;
}

// ---------------------------------------------------------------------------
// K7: x_proj GEMM  x_dbl[m,n] = sum_k xi[m,k]*W2[n,k]; M=32768, N=36(pad48), K=128.
// Tiles 2048x3 = 6144 -> 768 blocks of 8 waves.
// ---------------------------------------------------------------------------
__global__ void k_gemm_xproj(const _Float16* __restrict__ xih, const _Float16* __restrict__ w2_h,
                             float* __restrict__ xdbl) {
  const int lane = threadIdx.x & 31;
  const int tile = blockIdx.x * 8 + (threadIdx.x >> 5);
  const int mt = tile / 3;
  const int nt = tile % 3;
  const int mrow = mt * 16 + (lane & 15);
  const int col = nt * 16 + (lane & 15);
  v8f acc = vzero8();
#pragma unroll
  for (int kc = 0; kc < 4; ++kc) {
    const int kb = kc * 32;
    const _Float16* ap = xih + (size_t)mrow * 128 + kb + ((lane & 16) ? 8 : 0);
    v16h a = ld8x2(ap, ap + 16);
    const _Float16* bp = w2_h + (size_t)col * 128 + kb + ((lane & 16) ? 8 : 0);
    v16h bm = ld8x2(bp, bp + 16);
    acc = wmma16(a, bm, acc);
  }
  if (col < 36) {
    const int rowbase = mt * 16 + ((lane & 16) ? 8 : 0);
#pragma unroll
    for (int r = 0; r < 8; ++r)
      xdbl[(size_t)(rowbase + r) * 36 + col] = acc[r];
  }
}

// ---------------------------------------------------------------------------
// K8: selective scan. One block per bf (32 blocks, 128 threads = one di each).
// Per step: stage x_dbl row (dt4|B16|C16) in LDS, compute delta inline
// (dt_proj + softplus), update 16-wide state in registers, fuse SiLU-gate,
// emit g = ys * silu(z) as f16 for the out_proj WMMA.
// ---------------------------------------------------------------------------
__global__ void k_scan(const float* __restrict__ xdbl, const float* __restrict__ xic,
                       const float* __restrict__ z, const float* __restrict__ A_log,
                       const float* __restrict__ dt_w, const float* __restrict__ dt_b,
                       const float* __restrict__ D_skip, _Float16* __restrict__ g_h) {
  __shared__ float row[36];
  const int bf = blockIdx.x;
  const int di = threadIdx.x;
  float Arow[16];
#pragma unroll
  for (int s = 0; s < 16; ++s) Arow[s] = -__expf(A_log[di * 16 + s]);
  const float w0 = dt_w[di * 4 + 0], w1 = dt_w[di * 4 + 1];
  const float w2 = dt_w[di * 4 + 2], w3 = dt_w[di * 4 + 3];
  const float bia = dt_b[di], dsk = D_skip[di];
  float h[16];
#pragma unroll
  for (int s = 0; s < 16; ++s) h[s] = 0.f;

  for (int l = 0; l < 1024; ++l) {
    if (di < 36) row[di] = xdbl[((size_t)bf * 1024 + l) * 36 + di];
    __syncthreads();
    const float pre = row[0] * w0 + row[1] * w1 + row[2] * w2 + row[3] * w3 + bia;
    const float delta = (pre > 20.f) ? pre : log1pf(__expf(pre));
    const size_t idx = ((size_t)bf * 1024 + l) * 128 + di;
    const float xv = xic[idx];
    float yv = 0.f;
#pragma unroll
    for (int s = 0; s < 16; ++s) {
      const float dA = __expf(delta * Arow[s]);
      h[s] = dA * h[s] + delta * row[4 + s] * xv;
      yv += h[s] * row[20 + s];
    }
    yv += xv * dsk;
    const float zv = z[idx];
    const float g = yv * (zv / (1.f + __expf(-zv)));
    g_h[idx] = (_Float16)g;
    __syncthreads();
  }
}

// ---------------------------------------------------------------------------
// K9: out_proj GEMM out[m,n] = sum_k g[m,k]*W3[n,k]; M=32768, N=64, K=128.
// Epilogue scatters straight into output layout (B, C, H, W, Dep).
// Tiles 2048x4 = 8192 -> 1024 blocks of 8 waves.
// ---------------------------------------------------------------------------
__global__ void k_gemm_outproj(const _Float16* __restrict__ g_h, const _Float16* __restrict__ w3_h,
                               float* __restrict__ out) {
  const int lane = threadIdx.x & 31;
  const int tile = blockIdx.x * 8 + (threadIdx.x >> 5);
  const int mt = tile >> 2;
  const int nt = tile & 3;
  const int mrow = mt * 16 + (lane & 15);
  const int col = nt * 16 + (lane & 15);
  v8f acc = vzero8();
#pragma unroll
  for (int kc = 0; kc < 4; ++kc) {
    const int kb = kc * 32;
    const _Float16* ap = g_h + (size_t)mrow * 128 + kb + ((lane & 16) ? 8 : 0);
    v16h a = ld8x2(ap, ap + 16);
    const _Float16* bp = w3_h + (size_t)col * 128 + kb + ((lane & 16) ? 8 : 0);
    v16h bm = ld8x2(bp, bp + 16);
    acc = wmma16(a, bm, acc);
  }
  const int rowbase = mt * 16 + ((lane & 16) ? 8 : 0);
#pragma unroll
  for (int r = 0; r < 8; ++r) {
    const int m = rowbase + r;
    const int bf = m >> 10, l = m & 1023;
    const int b = bf >> 4, dep = bf & 15;
    const int hh = l >> 5, ww = l & 31;
    out[((((size_t)b * 64 + col) * 32 + hh) * 32 + ww) * 16 + dep] = acc[r];
  }
}

// ---------------------------------------------------------------------------
extern "C" void kernel_launch(void* const* d_in, const int* in_sizes, int n_in,
                              void* d_out, int out_size, void* d_ws, size_t ws_size,
                              hipStream_t stream) {
  (void)in_sizes; (void)n_in; (void)out_size; (void)ws_size;
  const float* x         = (const float*)d_in[0];
  const float* conv_w    = (const float*)d_in[1];
  const float* bn_gamma  = (const float*)d_in[2];
  const float* bn_beta   = (const float*)d_in[3];
  const float* in_proj_w = (const float*)d_in[4];
  const float* conv1d_w  = (const float*)d_in[5];
  const float* conv1d_b  = (const float*)d_in[6];
  const float* x_proj_w  = (const float*)d_in[7];
  const float* dt_proj_w = (const float*)d_in[8];
  const float* dt_proj_b = (const float*)d_in[9];
  const float* A_log     = (const float*)d_in[10];
  const float* D_skip    = (const float*)d_in[11];
  const float* out_proj_w= (const float*)d_in[12];
  float* out = (float*)d_out;

  char* base = (char*)d_ws;
  size_t off = 0;
  auto take = [&](size_t bytes) -> void* {
    void* p = base + off;
    off = (off + bytes + 255) & ~(size_t)255;
    return p;
  };
  _Float16* x_h    = (_Float16*)take(1048576 * 2);   // x in f16
  _Float16* wc_h   = (_Float16*)take(55296 * 2);     // conv w [tap][cout][cin]
  _Float16* w1_h   = (_Float16*)take(16384 * 2);     // in_proj (256x64)
  _Float16* w2_h   = (_Float16*)take(6144 * 2);      // x_proj padded (48x128)
  _Float16* w3_h   = (_Float16*)take(8192 * 2);      // out_proj (64x128)
  float* ch_sum    = (float*)take(64 * 4);
  float* ch_sumsq  = (float*)take(64 * 4);
  float* scale     = (float*)take(64 * 4);
  float* shift     = (float*)take(64 * 4);
  float* y_buf     = (float*)take(2097152 * 4);      // conv out, u-layout
  _Float16* u_h    = (_Float16*)take(2097152 * 2);   // BN+ReLU f16
  float* xi        = (float*)take(4194304 * 4);      // (32768,128)
  float* z_buf     = (float*)take(4194304 * 4);
  float* xic       = (float*)take(4194304 * 4);      // conv1d+silu f32
  _Float16* xih    = (_Float16*)take(4194304 * 2);   // conv1d+silu f16
  float* xdbl      = (float*)take(1179648 * 4);      // (32768,36)
  _Float16* g_h    = (_Float16*)take(4194304 * 2);   // ys*silu(z) f16

  k_pack<<<4096, 256, 0, stream>>>(x, conv_w, in_proj_w, x_proj_w, out_proj_w,
                                   x_h, wc_h, w1_h, w2_h, w3_h, ch_sum, ch_sumsq);
  k_conv_wmma<<<1024, 256, 0, stream>>>(x_h, wc_h, y_buf, ch_sum, ch_sumsq);
  k_bnfin<<<1, 64, 0, stream>>>(ch_sum, ch_sumsq, bn_gamma, bn_beta, scale, shift);
  k_bnrelu<<<8192, 256, 0, stream>>>(y_buf, scale, shift, u_h);
  k_gemm_inproj<<<4096, 256, 0, stream>>>(u_h, w1_h, xi, z_buf);
  k_conv1d<<<16384, 256, 0, stream>>>(xi, conv1d_w, conv1d_b, xic, xih);
  k_gemm_xproj<<<768, 256, 0, stream>>>(xih, w2_h, xdbl);
  k_scan<<<32, 128, 0, stream>>>(xdbl, xic, z_buf, A_log, dt_proj_w, dt_proj_b,
                                 D_skip, g_h);
  k_gemm_outproj<<<1024, 256, 0, stream>>>(g_h, w3_h, out);
}
